// SceneDecoder_2388001817173
// MI455X (gfx1250) — matrix operand order repaired and checked by reading, not compiled, not run on hardware
//
#include <hip/hip_runtime.h>

// ---------------- model constants ----------------
static constexpr int CB = 4, CN = 48, CL = 112, CS = 160, CM = 6, CT = 60;
static constexpr int CNM  = CN * CM;        // 288
static constexpr int CBNM = CB * CN * CM;   // 1152
static constexpr float CNEG = -1e9f;

typedef __attribute__((ext_vector_type(16))) _Float16 v16h;
typedef __attribute__((ext_vector_type(8)))  float    v8f;
typedef __attribute__((ext_vector_type(4)))  float    v4f;

// =================================================================
// Weight converter: W (f32, [K,N] row-major, or [N,K] if trans) ->
// f16 [Nout][K] row-major so the GEMM B operand is contiguous.
// =================================================================
__global__ void k_cvt_w(const float* __restrict__ W, int K, int N, int trans,
                        _Float16* __restrict__ out) {
  int id = blockIdx.x * 256 + threadIdx.x;
  if (id >= K * N) return;
  int n = id / K, k = id - n * K;
  float v = trans ? W[(size_t)n * K + k] : W[(size_t)k * N + n];
  out[id] = (_Float16)v;
}

// =================================================================
// WMMA GEMM: out[R,Nout] = A[R,K](f32) @ Wh ([Nout,K] f16) + bias
//            (+ residual) (opt relu).  f16 operands, f32 accumulate.
// One wave32 computes a 32x32 output macro-tile = 2x2 WMMA tiles
// sharing A/B operands -> 4 v_wmma per 32-wide K step.
// K must be a multiple of 32 (true at every call site: 128 / 512).
// No LDS, no per-element guards: OOB rows/cols are handled by
// clamping addresses (garbage lands only in C rows/cols that the
// guarded store discards).
// Operand packing (contiguous K, same for A and B; not the ISA table order):
//   A: lane L holds row (L&15); half i -> K = i + hi
//      with hi = 16*(L>=16)
//   B: lane L holds col (L&15); half i -> K = i + 16*(L>=16)
//   C/D: VGPR r -> row r + 8*(L>=16), col = L&15
// =================================================================
__global__ __launch_bounds__(32)
void k_gemm(const float* __restrict__ A, int lda,
            const _Float16* __restrict__ Wh,
            const float* __restrict__ bias, const float* __restrict__ res,
            float* __restrict__ out, int ldc,
            int R, int K, int Nout, int relu) {
  const int lane = threadIdx.x;
  const int row0 = blockIdx.x * 32;
  const int col0 = blockIdx.y * 32;
  const int rowl = lane & 15;
  const int hi   = (lane >= 16) ? 16 : 0;    // A K-phase
  const int kb   = (lane >= 16) ? 16 : 0;    // B K-phase

  const int ar0 = (row0 + rowl < R) ? (row0 + rowl) : (R - 1);
  const int ar1 = (row0 + 16 + rowl < R) ? (row0 + 16 + rowl) : (R - 1);
  const float* __restrict__ Arow0 = A + (size_t)ar0 * lda;
  const float* __restrict__ Arow1 = A + (size_t)ar1 * lda;

  const int n0 = col0 + rowl;
  const int n1 = col0 + 16 + rowl;
  const _Float16* __restrict__ Wr0 = Wh + (size_t)((n0 < Nout) ? n0 : (Nout - 1)) * K;
  const _Float16* __restrict__ Wr1 = Wh + (size_t)((n1 < Nout) ? n1 : (Nout - 1)) * K;

  v8f acc00 = {}, acc01 = {}, acc10 = {}, acc11 = {};
  for (int k0 = 0; k0 < K; k0 += 32) {
    if (k0 + 32 < K) {
      __builtin_prefetch(Arow0 + k0 + 32);
      __builtin_prefetch(Arow1 + k0 + 32);
    }
    // ---- A operands: 16 halfs per lane per row-tile, 4 x b128 + cvt ----
    const v4f a0 = *(const v4f*)(Arow0 + k0 + hi);
    const v4f a1 = *(const v4f*)(Arow0 + k0 + hi + 4);
    const v4f a2 = *(const v4f*)(Arow0 + k0 + hi + 8);
    const v4f a3 = *(const v4f*)(Arow0 + k0 + hi + 12);
    const v4f a4 = *(const v4f*)(Arow1 + k0 + hi);
    const v4f a5 = *(const v4f*)(Arow1 + k0 + hi + 4);
    const v4f a6 = *(const v4f*)(Arow1 + k0 + hi + 8);
    const v4f a7 = *(const v4f*)(Arow1 + k0 + hi + 12);
    v16h av0, av1;
#pragma unroll
    for (int j = 0; j < 4; ++j) {
      av0[j]      = (_Float16)a0[j];
      av0[4 + j]  = (_Float16)a1[j];
      av0[8 + j]  = (_Float16)a2[j];
      av0[12 + j] = (_Float16)a3[j];
      av1[j]      = (_Float16)a4[j];
      av1[4 + j]  = (_Float16)a5[j];
      av1[8 + j]  = (_Float16)a6[j];
      av1[12 + j] = (_Float16)a7[j];
    }
    // ---- B operands: 16 contiguous f16 per lane per col-tile ----
    const v16h bv0 = *(const v16h*)(Wr0 + k0 + kb);
    const v16h bv1 = *(const v16h*)(Wr1 + k0 + kb);

    acc00 = __builtin_amdgcn_wmma_f32_16x16x32_f16(false, av0, false, bv0,
                                                   (short)0, acc00, false, false);
    acc01 = __builtin_amdgcn_wmma_f32_16x16x32_f16(false, av0, false, bv1,
                                                   (short)0, acc01, false, false);
    acc10 = __builtin_amdgcn_wmma_f32_16x16x32_f16(false, av1, false, bv0,
                                                   (short)0, acc10, false, false);
    acc11 = __builtin_amdgcn_wmma_f32_16x16x32_f16(false, av1, false, bv1,
                                                   (short)0, acc11, false, false);
  }
  // ---- guarded epilogue / store ----
  const int rb = (lane >= 16) ? 8 : 0;
#pragma unroll
  for (int r = 0; r < 8; ++r) {
    int gr0 = row0 + rb + r;
    int gr1 = row0 + 16 + rb + r;
    if (gr0 < R) {
      if (n0 < Nout) {
        float v = acc00[r];
        if (bias) v += bias[n0];
        if (res)  v += res[(size_t)gr0 * ldc + n0];
        if (relu) v = fmaxf(v, 0.f);
        out[(size_t)gr0 * ldc + n0] = v;
      }
      if (n1 < Nout) {
        float v = acc01[r];
        if (bias) v += bias[n1];
        if (res)  v += res[(size_t)gr0 * ldc + n1];
        if (relu) v = fmaxf(v, 0.f);
        out[(size_t)gr0 * ldc + n1] = v;
      }
    }
    if (gr1 < R) {
      if (n0 < Nout) {
        float v = acc10[r];
        if (bias) v += bias[n0];
        if (res)  v += res[(size_t)gr1 * ldc + n0];
        if (relu) v = fmaxf(v, 0.f);
        out[(size_t)gr1 * ldc + n0] = v;
      }
      if (n1 < Nout) {
        float v = acc11[r];
        if (bias) v += bias[n1];
        if (res)  v += res[(size_t)gr1 * ldc + n1];
        if (relu) v = fmaxf(v, 0.f);
        out[(size_t)gr1 * ldc + n1] = v;
      }
    }
  }
}

// ---------------- LayerNorm over D=128, optional relu ----------------
__global__ void k_ln(const float* __restrict__ x, const float* __restrict__ g,
                     const float* __restrict__ b, float* __restrict__ o,
                     int R, int relu) {
  int row = blockIdx.x;
  int t = threadIdx.x;            // blockDim = 128
  if (row >= R) return;
  float v = x[(size_t)row * 128 + t];
  __shared__ float red[128];
  red[t] = v; __syncthreads();
  for (int s = 64; s > 0; s >>= 1) { if (t < s) red[t] += red[t + s]; __syncthreads(); }
  float mean = red[0] / 128.f; __syncthreads();
  float d = v - mean;
  red[t] = d * d; __syncthreads();
  for (int s = 64; s > 0; s >>= 1) { if (t < s) red[t] += red[t + s]; __syncthreads(); }
  float var = red[0] / 128.f;
  float y = d * rsqrtf(var + 1e-5f) * g[t] + b[t];
  if (relu) y = fmaxf(y, 0.f);
  o[(size_t)row * 128 + t] = y;
}

// ---------------- attention (wave per (batch,query,head)) ----------------
// mode: 0=cross (q->agent q/M, keys=scene, mask m1&m2-AND), 1=self (b=be/M),
//       2=none, 3=scored (!x_scored -> NEG)
__global__ __launch_bounds__(32)
void k_attn(const float* __restrict__ qh, const float* __restrict__ kh,
            const float* __restrict__ vh,
            const float* __restrict__ rk, const float* __restrict__ rv,
            const int* __restrict__ m1, const int* __restrict__ m2,
            float* __restrict__ out, int Qn, int Kn, int mode) {
  int wid = blockIdx.x;
  int h  = wid & 7;
  int q  = (wid >> 3) % Qn;
  int be = wid / (8 * Qn);
  int lane = threadIdx.x;
  int qrow = be * Qn + q;
  float qv[16];
#pragma unroll
  for (int d = 0; d < 16; ++d) qv[d] = qh[(size_t)qrow * 128 + h * 16 + d];

  float sv[5];
#pragma unroll
  for (int kk = 0; kk < 5; ++kk) {
    int k = lane + kk * 32;
    float s = -3.0e38f;
    if (k < Kn) {
      const float* kp = kh + (size_t)(be * Kn + k) * 128 + h * 16;
      float acc = 0.f;
#pragma unroll
      for (int d = 0; d < 16; ++d) acc += qv[d] * kp[d];
      if (rk) {
        size_t rbi = (mode == 0)
            ? (size_t)((be * CN + q / CM) * Kn + k) * 128 + h * 16
            : (size_t)(((be / CM) * CN + q) * Kn + k) * 128 + h * 16;
        const float* rp = rk + rbi;
#pragma unroll
        for (int d = 0; d < 16; ++d) acc += qv[d] * rp[d];
      }
      acc *= 0.25f;  // 1/sqrt(dh=16)
      if (mode == 0) {
        int mq = m1[be * CN + q / CM];
        int mk = (k < CN) ? m1[be * CN + k] : m2[be * (Kn - CN) + (k - CN)];
        if (mq && mk) acc += CNEG;
      } else if (mode == 1) {
        if (m1[(be / CM) * CN + k]) acc += CNEG;
      } else if (mode == 3) {
        if (!m1[(be / CM) * CN + k]) acc += CNEG;
      }
      s = acc;
    }
    sv[kk] = s;
  }
  float mx = sv[0];
#pragma unroll
  for (int kk = 1; kk < 5; ++kk) mx = fmaxf(mx, sv[kk]);
  for (int off = 16; off > 0; off >>= 1) mx = fmaxf(mx, __shfl_xor(mx, off));
  float sum = 0.f;
#pragma unroll
  for (int kk = 0; kk < 5; ++kk) {
    int k = lane + kk * 32;
    float e = (k < Kn) ? __expf(sv[kk] - mx) : 0.f;
    sv[kk] = e; sum += e;
  }
  for (int off = 16; off > 0; off >>= 1) sum += __shfl_xor(sum, off);
  float inv = 1.f / sum;
  float o[16];
#pragma unroll
  for (int d = 0; d < 16; ++d) o[d] = 0.f;
#pragma unroll
  for (int kk = 0; kk < 5; ++kk) {
    int k = lane + kk * 32;
    if (k < Kn) {
      float a = sv[kk] * inv;
      const float* vp = vh + (size_t)(be * Kn + k) * 128 + h * 16;
      if (rv) {
        size_t rbi = (mode == 0)
            ? (size_t)((be * CN + q / CM) * Kn + k) * 128 + h * 16
            : (size_t)(((be / CM) * CN + q) * Kn + k) * 128 + h * 16;
        const float* rp = rv + rbi;
#pragma unroll
        for (int d = 0; d < 16; ++d) o[d] += a * (vp[d] + rp[d]);
      } else {
#pragma unroll
        for (int d = 0; d < 16; ++d) o[d] += a * vp[d];
      }
    }
  }
  for (int off = 16; off > 0; off >>= 1) {
#pragma unroll
    for (int d = 0; d < 16; ++d) o[d] += __shfl_xor(o[d], off);
  }
  if (lane < 16) out[(size_t)qrow * 128 + h * 16 + lane] = o[lane];
}

// ---------------- small elementwise / misc kernels ----------------
__global__ void k_mlp5(const float* __restrict__ x, const float* __restrict__ w1,
                       const float* __restrict__ b1, float* __restrict__ out, int R) {
  int id = blockIdx.x * 256 + threadIdx.x;
  if (id >= R * 512) return;
  int col = id & 511, row = id >> 9;
  float acc = b1[col];
#pragma unroll
  for (int j = 0; j < 5; ++j) acc += x[(size_t)row * 5 + j] * w1[j * 512 + col];
  out[id] = fmaxf(acc, 0.f);
}

__global__ void k_init_tq(const float* __restrict__ atq, const float* __restrict__ scene,
                          float* __restrict__ tq) {
  int id = blockIdx.x * 256 + threadIdx.x;
  if (id >= CBNM * 128) return;
  int d = id & 127, r = id >> 7;
  int m = r % CM, n = (r / CM) % CN, b = r / (CM * CN);
  tq[id] = atq[m * 128 + d] + scene[(size_t)(b * CS + n) * 128 + d];
}

__global__ void k_perm_nm2mn(const float* __restrict__ in, float* __restrict__ out) {
  int id = blockIdx.x * 256 + threadIdx.x;
  if (id >= CBNM * 128) return;
  int d = id & 127, r = id >> 7;
  int m = r % CM, n = (r / CM) % CN, b = r / (CM * CN);
  out[((size_t)((b * CM + m) * CN + n) << 7) + d] = in[id];
}
__global__ void k_perm_mn2nm(const float* __restrict__ in, float* __restrict__ out) {
  int id = blockIdx.x * 256 + threadIdx.x;
  if (id >= CBNM * 128) return;
  int d = id & 127, r = id >> 7;
  int n = r % CN, m = (r / CN) % CM, b = r / (CM * CN);
  out[((size_t)((b * CN + n) * CM + m) << 7) + d] = in[id];
}

__global__ void k_maskzero(float* __restrict__ tq, const int* __restrict__ kpm) {
  int id = blockIdx.x * 256 + threadIdx.x;
  if (id >= CBNM * 128) return;
  if (kpm[(id >> 7) / CM]) tq[id] = 0.f;
}

__global__ void k_cumsum(float* __restrict__ loc) {
  int id = blockIdx.x * 256 + threadIdx.x;
  if (id >= CBNM * 2) return;
  int row = id >> 1, c = id & 1;
  float acc = 0.f;
  for (int t = 0; t < CT; ++t) {
    size_t p = (size_t)(row * CT + t) * 2 + c;
    acc += loc[p]; loc[p] = acc;
  }
}

__global__ void k_fourier(const float* __restrict__ y, const float* __restrict__ freqs,
                          const float* __restrict__ w1, const float* __restrict__ b1,
                          float* __restrict__ hcd) {
  int row = blockIdx.x;                   // BNM*T rows
  int c = blockIdx.y;                     // channel 0/1
  int t = threadIdx.x;                    // 128
  __shared__ float f[129];
  float xv = y[(size_t)row * 2 + c];
  if (t < 64) {
    float a = xv * freqs[c * 64 + t] * 6.283185307179586f;
    f[t]      = cosf(a);
    f[64 + t] = sinf(a);
  }
  if (t == 64) f[128] = xv;
  __syncthreads();
  const float* W = w1 + (size_t)c * 129 * 128;
  float acc = b1[c * 128 + t];
  for (int j = 0; j < 129; ++j) acc += f[j] * W[(size_t)j * 128 + t];
  hcd[((size_t)row * 2 + c) * 128 + t] = acc;
}

__global__ void k_bcast_h0(const float* __restrict__ h0, float* __restrict__ h) {
  int id = blockIdx.x * 256 + threadIdx.x;
  if (id >= CBNM * 128) return;
  h[id] = h0[id & 127];
}

__global__ void k_gru_gate(const float* __restrict__ gi, const float* __restrict__ gh,
                           float* __restrict__ h) {
  int id = blockIdx.x * 256 + threadIdx.x;
  if (id >= CBNM * 128) return;
  int row = id >> 7, d = id & 127;
  size_t b0 = (size_t)row * 384 + d;
  float ir = gi[b0],        hr = gh[b0];
  float iz = gi[b0 + 128],  hz = gh[b0 + 128];
  float in = gi[b0 + 256],  hn = gh[b0 + 256];
  float r = 1.f / (1.f + __expf(-(ir + hr)));
  float z = 1.f / (1.f + __expf(-(iz + hz)));
  float n = tanhf(in + r * hn);
  h[id] = (1.f - z) * n + z * h[id];
}

__global__ void k_init_sq(const float* __restrict__ sqw, float* __restrict__ sq) {
  int id = blockIdx.x * 256 + threadIdx.x;
  if (id >= CB * CM * 128) return;
  int m = (id >> 7) % CM;
  sq[id] = sqw[m * 128 + (id & 127)];
}

// =================================================================
// host side
// =================================================================
namespace {

struct Blk {
  const float *ln1g, *ln1b, *ln2g, *ln2b, *lnkvg, *lnkvb, *lnrg, *lnrb;
  const float *wq, *bq, *wk, *bk, *wv, *bv, *wo, *bo, *wrk, *wrv;
  const float *fw1, *fb1, *fw2, *fb2;
  const float *rw1, *rb1, *rw2, *rb2;
  bool rpe, upd, cross;
};

struct WS {
  float *qn, *kn, *qh, *kh, *vh, *att, *ffn, *rln, *rk, *rv;
};

// per-launch f16 weight converter (bump-allocates from the arena)
struct Cvt {
  hipStream_t st;
  char* base;
  size_t* off;
  const _Float16* operator()(const float* W, int K, int N, int trans) const {
    _Float16* p = (_Float16*)(base + *off);
    size_t bytes = ((size_t)K * N * sizeof(_Float16) + 255) & ~(size_t)255;
    *off += bytes;
    int total = K * N;
    k_cvt_w<<<(total + 255) / 256, 256, 0, st>>>(W, K, N, trans, p);
    return p;
  }
};

inline void gemm(hipStream_t st, const float* A, int lda, const _Float16* Wh,
                 const float* bias, const float* res, float* out,
                 int ldc, int R, int K, int Nout, int relu) {
  dim3 g((R + 31) / 32, (Nout + 31) / 32);
  k_gemm<<<g, 32, 0, st>>>(A, lda, Wh, bias, res, out, ldc, R, K, Nout, relu);
}
inline void lnorm(hipStream_t st, const float* x, const float* g, const float* b,
                  float* o, int R, int relu) {
  k_ln<<<R, 128, 0, st>>>(x, g, b, o, R, relu);
}

void run_block(const Cvt& cv, const Blk& p, float* x, const float* mem,
               int Be, int Qn, int Kn, float* rpe, int rpeRows, int mode,
               const int* m1, const int* m2, const WS& w) {
  hipStream_t st = cv.st;
  int Rq = Be * Qn, Rk = Be * Kn;
  // convert this block's weights to f16 [N,K]
  const _Float16* wq = cv(p.wq, 128, 128, 0);
  const _Float16* wk = cv(p.wk, 128, 128, 0);
  const _Float16* wv = cv(p.wv, 128, 128, 0);
  const _Float16* wo = cv(p.wo, 128, 128, 0);
  const _Float16* fw1 = cv(p.fw1, 128, 512, 0);
  const _Float16* fw2 = cv(p.fw2, 512, 128, 0);
  const _Float16 *wrk = nullptr, *wrv = nullptr, *rw1 = nullptr, *rw2 = nullptr;
  if (rpe && p.rpe) { wrk = cv(p.wrk, 128, 128, 0); wrv = cv(p.wrv, 128, 128, 0); }
  if (p.upd && rpe) { rw1 = cv(p.rw1, 128, 128, 0); rw2 = cv(p.rw2, 128, 128, 0); }

  lnorm(st, x, p.ln1g, p.ln1b, w.qn, Rq, 0);
  const float* knb = w.qn;
  if (mem != (const float*)x) {
    lnorm(st, mem, p.cross ? p.lnkvg : p.ln1g, p.cross ? p.lnkvb : p.ln1b, w.kn, Rk, 0);
    knb = w.kn;
  }
  gemm(st, w.qn, 128, wq, p.bq, nullptr, w.qh, 128, Rq, 128, 128, 0);
  gemm(st, knb, 128, wk, p.bk, nullptr, w.kh, 128, Rk, 128, 128, 0);
  gemm(st, knb, 128, wv, p.bv, nullptr, w.vh, 128, Rk, 128, 128, 0);
  if (rpe && p.rpe) {
    gemm(st, rpe, 128, wrk, nullptr, nullptr, w.rk, 128, rpeRows, 128, 128, 0);
    gemm(st, rpe, 128, wrv, nullptr, nullptr, w.rv, 128, rpeRows, 128, 128, 0);
  }
  k_attn<<<Be * Qn * 8, 32, 0, st>>>(w.qh, w.kh, w.vh,
                                     (rpe && p.rpe) ? w.rk : nullptr,
                                     (rpe && p.rpe) ? w.rv : nullptr,
                                     m1, m2, w.att, Qn, Kn, mode);
  gemm(st, w.att, 128, wo, p.bo, x, x, 128, Rq, 128, 128, 0);
  lnorm(st, x, p.ln2g, p.ln2b, w.qn, Rq, 0);
  gemm(st, w.qn, 128, fw1, p.fb1, nullptr, w.ffn, 512, Rq, 128, 512, 1);
  gemm(st, w.ffn, 512, fw2, p.fb2, x, x, 128, Rq, 512, 128, 0);
  if (p.upd && rpe) {
    lnorm(st, rpe, p.lnrg, p.lnrb, w.rln, rpeRows, 0);
    gemm(st, w.rln, 128, rw1, p.rb1, nullptr, w.rk, 128, rpeRows, 128, 128, 1);
    gemm(st, w.rk, 128, rw2, p.rb2, rpe, rpe, 128, rpeRows, 128, 128, 0);
  }
}

}  // namespace

extern "C" void kernel_launch(void* const* d_in, const int* in_sizes, int n_in,
                              void* d_out, int out_size, void* d_ws, size_t ws_size,
                              hipStream_t stream) {
  (void)in_sizes; (void)out_size; (void)ws_size;
  hipStream_t st = stream;

  // ---------------- inputs ----------------
  const float* scene_feat = (const float*)d_in[0];   // [B,S,D]
  const float* x_rpe      = (const float*)d_in[2];   // [B,N,N,5]
  const float* x_srpe     = (const float*)d_in[3];   // [B,N,S,5]
  const int*   kpm        = (const int*)d_in[4];     // [B,N]
  const int*   lpm        = (const int*)d_in[5];     // [B,L]
  const int*   scored     = (const int*)d_in[6];     // [B,N]

  // ---------------- params (pytree leaves, alphabetical keys) ----------------
  int pi_ = 7;
  auto P = [&](void) -> const float* {
    const float* p = (pi_ < n_in) ? (const float*)d_in[pi_] : nullptr;
    ++pi_; return p;
  };
  auto load_block = [&](bool rpe, bool upd, bool cross) -> Blk {
    Blk b{}; b.rpe = rpe; b.upd = upd; b.cross = cross;
    // attn: bk,bo,bq,bv,wk,wo,wq,wv[,wrk,wrv]
    b.bk = P(); b.bo = P(); b.bq = P(); b.bv = P();
    b.wk = P(); b.wo = P(); b.wq = P(); b.wv = P();
    if (rpe) { b.wrk = P(); b.wrv = P(); }
    // ffn: b1,b2,w1,w2
    b.fb1 = P(); b.fb2 = P(); b.fw1 = P(); b.fw2 = P();
    // ln1: b,g ; ln2: b,g
    b.ln1b = P(); b.ln1g = P(); b.ln2b = P(); b.ln2g = P();
    if (cross) { b.lnkvb = P(); b.lnkvg = P(); }
    if (upd) {
      b.lnrb = P(); b.lnrg = P();
      b.rb1 = P(); b.rb2 = P(); b.rw1 = P(); b.rw2 = P();
    }
    return b;
  };

  const float* atq = P();                                     // agent_traj_query
  Blk m2m_prop = load_block(false, false, false);             // mode2mode_propose
  Blk m2m_ref  = load_block(false, false, false);             // mode2mode_refine
  const float *pd_b1 = P(), *pd_b2 = P(), *pd_w1 = P(), *pd_w2 = P();  // prob_decoder
  Blk prop_cross[2], prop_self[2], ref_cross[2], ref_self[2], s2m[2];
  for (int l = 0; l < 2; ++l) {                               // propose
    prop_cross[l] = load_block(true, true, true);
    prop_self[l]  = load_block(true, true, false);
  }
  ref_cross[0] = load_block(true, true, true);                // refine[0]
  ref_self[0]  = load_block(true, true, false);
  ref_cross[1] = load_block(true, false, true);               // refine[1]
  ref_self[1]  = load_block(true, false, false);
  for (int l = 0; l < 2; ++l) s2m[l] = load_block(false, false, true);  // scene_2_mode
  const float* scene_query = P();
  const float *tlp_b1 = P(), *tlp_b2 = P(), *tlp_w1 = P(), *tlp_w2 = P();
  const float *tlr_b1 = P(), *tlr_b2 = P(), *tlr_w1 = P(), *tlr_w2 = P();
  const float *ge_bhh = P(), *ge_bih = P(), *ge_whh = P(), *ge_wih = P();
  const float* h0 = P();
  const float *xp_b1 = P(), *xp_b2 = P(), *xp_w1 = P(), *xp_w2 = P();
  const float *xs_b1 = P(), *xs_b2 = P(), *xs_w1 = P(), *xs_w2 = P();
  const float *ye_b1 = P(), *ye_b2 = P(), *ye_b3 = P(), *ye_freqs = P();
  const float *ye_ln1b = P(), *ye_ln1g = P(), *ye_ln2b = P(), *ye_ln2g = P();
  const float *ye_w1 = P(), *ye_w2 = P(), *ye_w3 = P();

  // ---------------- outputs ----------------
  float* outp  = (float*)d_out;
  float* YHAT  = outp;                       // [B,N,M,T,2] = 138240
  float* PI    = outp + 138240;              // [B,M,1]     = 24
  float* YPROP = outp + 138264;              // [B,N,M,T,2] = 138240

  // ---------------- workspace arena ----------------
  char* base = (char*)d_ws;
  size_t off = 0;
  auto alloc = [&](size_t n) -> float* {
    float* p = (float*)(base + off);
    off += n * sizeof(float);
    return p;
  };
  float* TQ   = alloc((size_t)CBNM * 128);
  float* TQ2  = alloc((size_t)CBNM * 128);
  float* SRPE = alloc((size_t)CB * CN * CS * 128);     // shared across modes (dedup)
  float* ARPE = alloc((size_t)CB * CN * CN * 128);
  WS w;
  w.qn  = alloc((size_t)CBNM * 128);
  w.kn  = alloc((size_t)CBNM * 128);
  w.qh  = alloc((size_t)CBNM * 128);
  w.kh  = alloc((size_t)CBNM * 128);
  w.vh  = alloc((size_t)CBNM * 128);
  w.att = alloc((size_t)CBNM * 128);
  w.ffn = alloc((size_t)CBNM * 512);
  w.rln = alloc((size_t)CB * CN * CS * 128);
  w.rk  = alloc((size_t)CB * CN * CS * 128);
  w.rv  = alloc((size_t)CB * CN * CS * 128);
  float* BIG = alloc((size_t)CBNM * CT * 256);         // 17.7M f: mlp hidden / hcd
  float* HS  = alloc((size_t)CBNM * CT * 128);
  float* EMB = alloc((size_t)CBNM * CT * 128);
  float* GI  = alloc((size_t)CBNM * 384);
  float* GH  = alloc((size_t)CBNM * 384);
  float* SQ  = alloc((size_t)CB * CM * 128);
  Cvt cv{st, base, &off};   // f16 weights bump-allocated after the f32 buffers

  const int RSR = CB * CN * CS;   // 30720 rpe rows (scene)
  const int RAR = CB * CN * CN;   // 9216  rpe rows (agent)
  auto blks = [](int n) { return (unsigned)((n + 255) / 256); };

  // ---------------- RPE embeddings (mode-dedup: [B,N,*,D]) ----------------
  const _Float16* xsw2 = cv(xs_w2, 512, 128, 0);
  const _Float16* xpw2 = cv(xp_w2, 512, 128, 0);
  k_mlp5<<<blks(RSR * 512), 256, 0, st>>>(x_srpe, xs_w1, xs_b1, BIG, RSR);
  gemm(st, BIG, 512, xsw2, xs_b2, nullptr, SRPE, 128, RSR, 512, 128, 0);
  k_mlp5<<<blks(RAR * 512), 256, 0, st>>>(x_rpe, xp_w1, xp_b1, BIG, RAR);
  gemm(st, BIG, 512, xpw2, xp_b2, nullptr, ARPE, 128, RAR, 512, 128, 0);

  // tq init: agent_traj_query[m] + scene_feat[b,n]
  k_init_tq<<<blks(CBNM * 128), 256, 0, st>>>(atq, scene_feat, TQ);

  auto run_stage = [&](Blk* cross, Blk* self) {
    for (int l = 0; l < 2; ++l) {
      run_block(cv, cross[l], TQ, scene_feat, CB, CNM, CS, SRPE, RSR, 0, kpm, lpm, w);
      k_perm_nm2mn<<<blks(CBNM * 128), 256, 0, st>>>(TQ, TQ2);
      run_block(cv, self[l], TQ2, TQ2, CB * CM, CN, CN, ARPE, RAR, 1, kpm, nullptr, w);
      k_perm_mn2nm<<<blks(CBNM * 128), 256, 0, st>>>(TQ2, TQ);
    }
  };

  // ================= propose =================
  run_stage(prop_cross, prop_self);
  run_block(cv, m2m_prop, TQ, TQ, CB * CN, CM, CM, nullptr, 0, 2, nullptr, nullptr, w);
  k_maskzero<<<blks(CBNM * 128), 256, 0, st>>>(TQ, kpm);
  gemm(st, TQ, 128, cv(tlp_w1, 128, 512, 0), tlp_b1, nullptr, w.ffn, 512, CBNM, 128, 512, 1);
  gemm(st, w.ffn, 512, cv(tlp_w2, 512, 120, 0), tlp_b2, nullptr, YPROP, 120, CBNM, 512, 120, 0);
  k_cumsum<<<blks(CBNM * 2), 256, 0, st>>>(YPROP);

  // ================= refine: fourier + GRU =================
  const int RT = CBNM * CT;  // 69120
  k_fourier<<<dim3((unsigned)RT, 2), 128, 0, st>>>(YPROP, ye_freqs, ye_w1, ye_b1, BIG);
  lnorm(st, BIG, ye_ln1g, ye_ln1b, BIG, RT * 2, 1);
  gemm(st, BIG, 256, cv(ye_w2, 128, 128, 0), ye_b2, nullptr, HS, 128, RT, 128, 128, 0);
  gemm(st, BIG + 128, 256, cv(ye_w2 + 128 * 128, 128, 128, 0), ye_b2 + 128, HS, HS, 128,
       RT, 128, 128, 0);
  lnorm(st, HS, ye_ln2g, ye_ln2b, HS, RT, 1);
  gemm(st, HS, 128, cv(ye_w3, 128, 128, 0), ye_b3, nullptr, EMB, 128, RT, 128, 128, 0);

  const _Float16* wihH = cv(ge_wih, 128, 384, 1);   // [384,128] already [N,K]
  const _Float16* whhH = cv(ge_whh, 128, 384, 1);
  k_bcast_h0<<<blks(CBNM * 128), 256, 0, st>>>(h0, TQ);      // h lives in TQ
  for (int t = 0; t < CT; ++t) {
    gemm(st, EMB + (size_t)t * 128, CT * 128, wihH, ge_bih, nullptr, GI, 384,
         CBNM, 128, 384, 0);
    gemm(st, TQ, 128, whhH, ge_bhh, nullptr, GH, 384, CBNM, 128, 384, 0);
    k_gru_gate<<<blks(CBNM * 128), 256, 0, st>>>(GI, GH, TQ);
  }

  run_stage(ref_cross, ref_self);
  run_block(cv, m2m_ref, TQ, TQ, CB * CN, CM, CM, nullptr, 0, 2, nullptr, nullptr, w);
  k_maskzero<<<blks(CBNM * 128), 256, 0, st>>>(TQ, kpm);
  gemm(st, TQ, 128, cv(tlr_w1, 128, 512, 0), tlr_b1, nullptr, w.ffn, 512, CBNM, 128, 512, 1);
  gemm(st, w.ffn, 512, cv(tlr_w2, 512, 120, 0), tlr_b2, YPROP, YHAT, 120, CBNM, 512, 120, 0);

  // ================= scene -> mode scoring =================
  k_perm_nm2mn<<<blks(CBNM * 128), 256, 0, st>>>(TQ, TQ2);
  k_init_sq<<<blks(CB * CM * 128), 256, 0, st>>>(scene_query, SQ);
  for (int l = 0; l < 2; ++l)
    run_block(cv, s2m[l], SQ, TQ2, CB * CM, 1, CN, nullptr, 0, 3, scored, nullptr, w);
  gemm(st, SQ, 128, cv(pd_w1, 128, 512, 0), pd_b1, nullptr, w.ffn, 512, CB * CM, 128, 512, 1);
  gemm(st, w.ffn, 512, cv(pd_w2, 512, 1, 0), pd_b2, nullptr, PI, 1, CB * CM, 512, 1, 0);
}
